// res_block_12008728559853
// MI455X (gfx1250) — matrix operand order repaired and checked by reading, not compiled, not run on hardware
//
#include <hip/hip_runtime.h>

typedef unsigned short u16;
typedef unsigned int   u32;
typedef __attribute__((ext_vector_type(16))) __bf16 v16bf;
typedef __attribute__((ext_vector_type(8)))  __bf16 v8bf;
typedef __attribute__((ext_vector_type(8)))  float  v8f;

#define T_LEN   1024
#define B_N     64
#define CIN     448    // folded x channels
#define CH1     224    // hidden channels
#define CH2     896    // conv2 output channels
#define NSAMP   65536.0f
#define STRIDE1 456    // LDS row stride (bf16) for conv1 tile (228 dwords -> 16 conflict-free t-lanes)
#define STRIDE2 232    // LDS row stride (bf16) for conv2 tile (116 dwords)

__device__ __forceinline__ u16 f2bf(float f) {
    u32 u = __float_as_uint(f);
    u += 0x7FFFu + ((u >> 16) & 1u);          // round-to-nearest-even
    return (u16)(u >> 16);
}
__device__ __forceinline__ float bf2f(u16 h) {
    return __uint_as_float(((u32)h) << 16);
}
__device__ __forceinline__ int reflect_t(int t) {
    return (t < 0) ? -t : ((t > T_LEN - 1) ? 2 * (T_LEN - 1) - t : t);
}
__device__ __forceinline__ v16bf lds_frag(const u16* p) {
    union { v16bf v; v8bf h[2]; } u;
    u.h[0] = *(const v8bf*)(p);
    u.h[1] = *(const v8bf*)(p + 8);
    return u.v;
}

// ---------------- stats zero ----------------
__global__ void zero_stats(float* st) {
    for (int i = threadIdx.x; i < 2240; i += 256) st[i] = 0.0f;
}

// ---------------- weight prep: mask + unpool-fold + WMMA A-fragment pack ----------------
// A fragment packed in contiguous k order (B reads the same): lane l (r=l&15, half=l>>4) element i holds
// A[r][ half*16 + i ] within the 32-wide K block.
__global__ void prep_w1(const float* __restrict__ w1, const float* __restrict__ m1,
                        u16* __restrict__ w1f) {
    int e = blockIdx.x * 256 + threadIdx.x;
    if (e >= 3 * 14 * 14 * 512) return;
    int i = e & 15, lane = (e >> 4) & 31, f = e >> 9;
    int mt = f % 14, kb = (f / 14) % 14, tap = f / 196;
    int r = lane & 15, half = lane >> 4;
    int kk = (half << 4) + i;
    int cx = kb * 32 + kk;                 // folded channel 0..447
    int cout = mt * 16 + r;                // 0..223
    int p = cx >> 6, q = cx & 63;
    int cin0 = 128 * p + q, cin1 = cin0 + 64;
    float w = w1[(cout * 896 + cin0) * 3 + tap] * m1[cout * 896 + cin0]
            + w1[(cout * 896 + cin1) * 3 + tap] * m1[cout * 896 + cin1];
    w1f[e] = f2bf(w);
}

__global__ void prep_w2(const float* __restrict__ w2, const float* __restrict__ m2,
                        u16* __restrict__ w2f) {
    int e = blockIdx.x * 256 + threadIdx.x;
    if (e >= 3 * 7 * 56 * 512) return;
    int i = e & 15, lane = (e >> 4) & 31, f = e >> 9;
    int mt = f % 56, kb = (f / 56) % 7, tap = f / 392;
    int r = lane & 15, half = lane >> 4;
    int kk = (half << 4) + i;
    int cin = kb * 32 + kk;                // 0..223
    int cout = mt * 16 + r;                // 0..895
    w2f[e] = f2bf(w2[(cout * 224 + cin) * 3 + tap] * m2[cout * 224 + cin]);
}

// ---------------- conv1: x (folded) -> y1 (pre-BN, bf16) + BN1 partial stats ----------------
__global__ __launch_bounds__(448) void conv1_kernel(
    const float* __restrict__ x, const u16* __restrict__ w1f,
    const float* __restrict__ b1, u16* __restrict__ y1,
    float* __restrict__ sum1, float* __restrict__ sq1) {
    __shared__ u16 lds[66 * STRIDE1];
    const int b  = blockIdx.y;
    const int t0 = blockIdx.x * 64;
    const int tid = threadIdx.x;

    // Stage reflected [66 x 448] bf16 tile: thread = channel, loop rows (line reuse across iters)
    {
        const float* xc = x + ((size_t)b * CIN + tid) * T_LEN;
        for (int rt = 0; rt < 66; ++rt) {
            int t = reflect_t(t0 - 1 + rt);
            lds[rt * STRIDE1 + tid] = f2bf(xc[t]);
        }
    }
    __syncthreads();

    const int wave = tid >> 5, lane = tid & 31;
    const int lrow = lane & 15, lhalf = lane >> 4;
    v8f acc[4] = {};

    for (int tap = 0; tap < 3; ++tap) {
        for (int kb = 0; kb < 14; ++kb) {
            v16bf a = *(const v16bf*)(w1f + (((tap * 14 + kb) * 14 + wave) << 9) + lane * 16);
            int col = kb * 32 + (lhalf << 4);
            v16bf bm[4];
#pragma unroll
            for (int nf = 0; nf < 4; ++nf) {
                int row = nf * 16 + lrow + tap;
                bm[nf] = lds_frag(&lds[row * STRIDE1 + col]);
            }
#pragma unroll
            for (int nf = 0; nf < 4; ++nf) {
                acc[nf] = __builtin_amdgcn_wmma_f32_16x16x32_bf16(
                    false, a, false, bm[nf], (short)0, acc[nf], false, false);
            }
        }
    }

    // bias, store bf16, per-channel stats (each channel owned by one wave half-row)
    float s[8], q[8];
#pragma unroll
    for (int j = 0; j < 8; ++j) { s[j] = 0.0f; q[j] = 0.0f; }
#pragma unroll
    for (int j = 0; j < 8; ++j) {
        int cout = wave * 16 + lhalf * 8 + j;
        float bias = b1[cout];
        u16* yrow = y1 + ((size_t)b * CH1 + cout) * T_LEN + t0 + lrow;
#pragma unroll
        for (int nf = 0; nf < 4; ++nf) {
            float v = acc[nf][j] + bias;
            yrow[nf * 16] = f2bf(v);
            s[j] += v; q[j] += v * v;
        }
    }
#pragma unroll
    for (int j = 0; j < 8; ++j) {
        float sv = s[j], qv = q[j];
        for (int m = 1; m < 16; m <<= 1) {
            sv += __shfl_xor(sv, m, 32);
            qv += __shfl_xor(qv, m, 32);
        }
        if (lrow == 0) {
            int cout = wave * 16 + lhalf * 8 + j;
            atomicAdd(&sum1[cout], sv);
            atomicAdd(&sq1[cout], qv);
        }
    }
}

// ---------------- BN finalize: a = g*rsqrt(var+eps), c = be - mean*a ----------------
__global__ void bn_fin(const float* __restrict__ sum, const float* __restrict__ sq,
                       const float* __restrict__ g, const float* __restrict__ be,
                       float* __restrict__ a, float* __restrict__ c, int n) {
    int i = blockIdx.x * blockDim.x + threadIdx.x;
    if (i >= n) return;
    const float inv = 1.0f / NSAMP;
    float m = sum[i] * inv;
    float v = sq[i] * inv - m * m;
    float s = g[i] * rsqrtf(v + 1e-5f);
    a[i] = s;
    c[i] = be[i] - m * s;
}

// ---------------- conv2: h=relu(BN1(y1)) -> y2 (pre-BN, bf16) + BN2 partial stats ----------------
__global__ __launch_bounds__(448) void conv2_kernel(
    const u16* __restrict__ y1, const u16* __restrict__ w2f,
    const float* __restrict__ a1, const float* __restrict__ c1,
    const float* __restrict__ b2, u16* __restrict__ y2,
    float* __restrict__ sum2, float* __restrict__ sq2) {
    __shared__ u16 lds[66 * STRIDE2];
    const int b  = blockIdx.y;
    const int t0 = blockIdx.x * 64;
    const int mt = blockIdx.z;             // 0..3, cout base mt*224
    const int tid = threadIdx.x;

    // Stage reflected [66 x 224] h tile with BN1 affine + ReLU applied, bf16
    {
        int c = tid % 224, seg = tid / 224;
        const u16* hc = y1 + ((size_t)b * CH1 + c) * T_LEN;
        float av = a1[c], cv = c1[c];
        for (int rr = 0; rr < 33; ++rr) {
            int rt = seg * 33 + rr;
            int t = reflect_t(t0 - 1 + rt);
            float v = fmaxf(av * bf2f(hc[t]) + cv, 0.0f);
            lds[rt * STRIDE2 + c] = f2bf(v);
        }
    }
    __syncthreads();

    const int wave = tid >> 5, lane = tid & 31;
    const int lrow = lane & 15, lhalf = lane >> 4;
    v8f acc[4] = {};

    for (int tap = 0; tap < 3; ++tap) {
        for (int kb = 0; kb < 7; ++kb) {
            v16bf a = *(const v16bf*)(w2f + (((tap * 7 + kb) * 56 + (mt * 14 + wave)) << 9) + lane * 16);
            int col = kb * 32 + (lhalf << 4);
            v16bf bm[4];
#pragma unroll
            for (int nf = 0; nf < 4; ++nf) {
                int row = nf * 16 + lrow + tap;
                bm[nf] = lds_frag(&lds[row * STRIDE2 + col]);
            }
#pragma unroll
            for (int nf = 0; nf < 4; ++nf) {
                acc[nf] = __builtin_amdgcn_wmma_f32_16x16x32_bf16(
                    false, a, false, bm[nf], (short)0, acc[nf], false, false);
            }
        }
    }

    float s[8], q[8];
#pragma unroll
    for (int j = 0; j < 8; ++j) { s[j] = 0.0f; q[j] = 0.0f; }
#pragma unroll
    for (int j = 0; j < 8; ++j) {
        int cout = mt * 224 + wave * 16 + lhalf * 8 + j;
        float bias = b2[cout];
        u16* yrow = y2 + ((size_t)b * CH2 + cout) * T_LEN + t0 + lrow;
#pragma unroll
        for (int nf = 0; nf < 4; ++nf) {
            float v = acc[nf][j] + bias;
            yrow[nf * 16] = f2bf(v);
            s[j] += v; q[j] += v * v;
        }
    }
#pragma unroll
    for (int j = 0; j < 8; ++j) {
        float sv = s[j], qv = q[j];
        for (int m = 1; m < 16; m <<= 1) {
            sv += __shfl_xor(sv, m, 32);
            qv += __shfl_xor(qv, m, 32);
        }
        if (lrow == 0) {
            int cout = mt * 224 + wave * 16 + lhalf * 8 + j;
            atomicAdd(&sum2[cout], sv);
            atomicAdd(&sq2[cout], qv);
        }
    }
}

// ---------------- epilogue: BN2 affine + pool(pairs) + residual + ReLU ----------------
__global__ void epilogue_kernel(const u16* __restrict__ y2, const float* __restrict__ x,
                                const float* __restrict__ a2, const float* __restrict__ c2,
                                float* __restrict__ out) {
    int idx = blockIdx.x * 256 + threadIdx.x;        // over 64*448*256 float4 groups
    if (idx >= B_N * CIN * (T_LEN / 4)) return;
    int t4 = (idx & 255) * 4;
    int co = (idx >> 8) % CIN;
    int b  = idx / (256 * CIN);
    int s  = co >> 6, q = co & 63;
    int ch0 = 128 * s + q, ch1 = ch0 + 64;
    const u16* p0 = y2 + ((size_t)b * CH2 + ch0) * T_LEN + t4;
    const u16* p1 = y2 + ((size_t)b * CH2 + ch1) * T_LEN + t4;
    const float4 xv = *(const float4*)(x + ((size_t)b * CIN + co) * T_LEN + t4);
    float A0 = a2[ch0], C0 = c2[ch0], A1 = a2[ch1], C1 = c2[ch1];
    uint2 r0 = *(const uint2*)p0;
    uint2 r1 = *(const uint2*)p1;
    float y0[4] = { bf2f((u16)(r0.x & 0xffff)), bf2f((u16)(r0.x >> 16)),
                    bf2f((u16)(r0.y & 0xffff)), bf2f((u16)(r0.y >> 16)) };
    float y1[4] = { bf2f((u16)(r1.x & 0xffff)), bf2f((u16)(r1.x >> 16)),
                    bf2f((u16)(r1.y & 0xffff)), bf2f((u16)(r1.y >> 16)) };
    float xr[4] = { xv.x, xv.y, xv.z, xv.w };
    float o[4];
#pragma unroll
    for (int i = 0; i < 4; ++i) {
        float v = 0.5f * ((A0 * y0[i] + C0) + (A1 * y1[i] + C1)) + xr[i];
        o[i] = fmaxf(v, 0.0f);
    }
    *(float4*)(out + ((size_t)b * CIN + co) * T_LEN + t4) = make_float4(o[0], o[1], o[2], o[3]);
}

extern "C" void kernel_launch(void* const* d_in, const int* in_sizes, int n_in,
                              void* d_out, int out_size, void* d_ws, size_t ws_size,
                              hipStream_t stream) {
    const float* x    = (const float*)d_in[0];
    const float* w1   = (const float*)d_in[1];
    const float* b1   = (const float*)d_in[2];
    const float* g1   = (const float*)d_in[3];
    const float* be1  = (const float*)d_in[4];
    const float* w2   = (const float*)d_in[5];
    const float* b2   = (const float*)d_in[6];
    const float* g2   = (const float*)d_in[7];
    const float* be2  = (const float*)d_in[8];
    const float* m1   = (const float*)d_in[9];
    const float* m2   = (const float*)d_in[10];
    float* out = (float*)d_out;

    char* wsb = (char*)d_ws;
    u16* w1f = (u16*)wsb;                               //   602,112 B  (224*448*3 bf16)
    u16* w2f = (u16*)(wsb + 602112);                    // 1,204,224 B  (896*224*3 bf16)
    u16* y1  = (u16*)(wsb + 1806336);                   // 29,360,128 B (64*224*1024 bf16)
    u16* y2  = (u16*)(wsb + 31166464);                  // 117,440,512 B (64*896*1024 bf16)
    float* st = (float*)(wsb + 148606976);              // stats + affine params
    float* sum1 = st;           float* sq1 = st + 224;
    float* sum2 = st + 448;     float* sq2 = st + 1344;
    float* a1   = st + 2240;    float* c1  = st + 2464;
    float* a2   = st + 2688;    float* c2  = st + 3584;

    zero_stats<<<1, 256, 0, stream>>>(st);
    prep_w1<<<(3 * 14 * 14 * 512) / 256, 256, 0, stream>>>(w1, m1, w1f);
    prep_w2<<<(3 * 7 * 56 * 512) / 256, 256, 0, stream>>>(w2, m2, w2f);

    conv1_kernel<<<dim3(16, 64), 448, 0, stream>>>(x, w1f, b1, y1, sum1, sq1);
    bn_fin<<<1, 224, 0, stream>>>(sum1, sq1, g1, be1, a1, c1, 224);
    conv2_kernel<<<dim3(16, 64, 4), 448, 0, stream>>>(y1, w2f, a1, c1, b2, y2, sum2, sq2);
    bn_fin<<<4, 224, 0, stream>>>(sum2, sq2, g2, be2, a2, c2, 896);
    epilogue_kernel<<<(B_N * CIN * (T_LEN / 4)) / 256, 256, 0, stream>>>(y2, x, a2, c2, out);
}
